// MultiheadAttention_63952063037965
// MI455X (gfx1250) — matrix operand order repaired and checked by reading, not compiled, not run on hardware
//
#include <hip/hip_runtime.h>
#include <hip/hip_bf16.h>

typedef __attribute__((ext_vector_type(16))) __bf16 v16bf;
typedef __attribute__((ext_vector_type(8)))  float  v8f;
typedef __attribute__((ext_vector_type(4)))  unsigned int v4u;
typedef __attribute__((ext_vector_type(8)))  int  v8i;
typedef __attribute__((ext_vector_type(4)))  int  v4i;

#define B_  4
#define S_  2048
#define F_  1024
#define H_  16
#define D_  64
#define HD_ (H_ * D_)

#if defined(__AMDGCN__) && __has_builtin(__builtin_amdgcn_tensor_load_to_lds) && \
    __has_builtin(__builtin_amdgcn_s_wait_tensorcnt)
#define HAVE_TDM 1
#else
#define HAVE_TDM 0
#endif

// ---------------------------------------------------------------------------
// WMMA wrapper: D = A(16x32 bf16) * B(32x16 bf16) + C(16x16 f32)
// ---------------------------------------------------------------------------
static __device__ inline v8f wmma_bf16(v16bf a, v16bf b, v8f c) {
  return __builtin_amdgcn_wmma_f32_16x16x32_bf16(
      /*neg_a=*/false, a, /*neg_b=*/false, b,
      /*c_mod=*/(short)0, c, /*reuse_a=*/false, /*reuse_b=*/false);
}

// ---------------------------------------------------------------------------
// Fragment loaders (wave32); A and B share a contiguous k order (not the ISA table order).
// A 16x32 (bf16): lane m = lane&15; lanes<16 hold K 0..15; lanes>=16
// hold K 16..31 (2 elems per VGPR).
// B 32x16 (bf16): lane n = lane&15; lanes<16 hold K 0..15, lanes>=16 K 16..31.
// C 16x16 (f32):  lane n = lane&15; VGPR r holds M = r (+8 for lanes>=16).
// ---------------------------------------------------------------------------
static __device__ inline v16bf load_a_f32(const float* __restrict__ base, int lda) {
  const int lane = threadIdx.x & 31;
  const int m  = lane & 15;
  const int kb = (lane >> 4) << 4;                 // 0 or 16
  const float* p = base + (size_t)m * lda + kb;
  v16bf a;
#pragma unroll
  for (int i = 0; i < 8; ++i) a[i] = (__bf16)p[i];
#pragma unroll
  for (int i = 0; i < 8; ++i) a[8 + i] = (__bf16)p[8 + i];
  return a;
}

static __device__ inline v16bf load_a_bf16(const __bf16* base, int lda) {
  const int lane = threadIdx.x & 31;
  const int m  = lane & 15;
  const int kb = (lane >> 4) << 4;
  const __bf16* p = base + (size_t)m * lda + kb;
  v16bf a;
#pragma unroll
  for (int i = 0; i < 8; ++i) a[i] = p[i];
#pragma unroll
  for (int i = 0; i < 8; ++i) a[8 + i] = p[8 + i];
  return a;
}

// Row-major B source: B[k][n] = base[k*ldb + n]
static __device__ inline v16bf load_b_f32(const float* __restrict__ base, int ldb) {
  const int lane = threadIdx.x & 31;
  const int n  = lane & 15;
  const int k0 = (lane >> 4) << 4;                 // 0 or 16
  v16bf b;
#pragma unroll
  for (int i = 0; i < 16; ++i) b[i] = (__bf16)base[(size_t)(k0 + i) * ldb + n];
  return b;
}

static __device__ inline v16bf load_b_bf16(const __bf16* base, int ldb) {
  const int lane = threadIdx.x & 31;
  const int n  = lane & 15;
  const int k0 = (lane >> 4) << 4;
  v16bf b;
#pragma unroll
  for (int i = 0; i < 16; ++i) b[i] = base[(size_t)(k0 + i) * ldb + n];
  return b;
}

// Transposed B source (used for Q*K^T): B[k][n] = base[n*ld + k]
// => per-lane fully contiguous 16-element load.
static __device__ inline v16bf load_bT_bf16(const __bf16* base, int ld) {
  const int lane = threadIdx.x & 31;
  const int n  = lane & 15;
  const int k0 = (lane >> 4) << 4;
  const __bf16* p = base + (size_t)n * ld + k0;
  v16bf b;
#pragma unroll
  for (int i = 0; i < 16; ++i) b[i] = p[i];
  return b;
}

static __device__ inline void store_c_bf16(__bf16* base, int ldc, v8f c) {
  const int lane = threadIdx.x & 31;
  const int n  = lane & 15;
  const int r0 = (lane >> 4) << 3;
#pragma unroll
  for (int r = 0; r < 8; ++r) base[(size_t)(r0 + r) * ldc + n] = (__bf16)c[r];
}

static __device__ inline void store_c_f32(float* base, int ldc, v8f c) {
  const int lane = threadIdx.x & 31;
  const int n  = lane & 15;
  const int r0 = (lane >> 4) << 3;
#pragma unroll
  for (int r = 0; r < 8; ++r) base[(size_t)(r0 + r) * ldc + n] = c[r];
}

// ---------------------------------------------------------------------------
// Tensor Data Mover: async DMA of a 2-D bf16 tile (tile_d0 x tile_d1) from
// global memory into LDS.  Descriptor packed per CDNA5 ISA sec. 8.3 / 8.4:
//   group0: count=1 | lds_addr | global_addr[56:0] | type=2
//   group1: data_size=1 (2B) | tensor_dim0/1 | tile_dim0/1 | dim0 stride
// Tracked with TENSORcnt; EXEC ignored (whole-wave op, scalar operands).
// This toolchain exposes the 6-argument builtin form:
//   (uint32x4 g0, int32x8 g1, int32x4 g2, int32x4 g3, int32x8 g4, i32 cpol)
// ---------------------------------------------------------------------------
#if HAVE_TDM
static __device__ inline void tdm_load_2d_bf16(unsigned lds_off, const void* gptr,
                                               int tile_d0, int tile_d1,
                                               int tensor_d0, int tensor_d1,
                                               long long stride0) {
  const unsigned long long ga = (unsigned long long)(uintptr_t)gptr;
  v4u g0;
  g0.x = 1u;                                              // count=1 (valid D#)
  g0.y = lds_off;                                         // LDS byte address
  g0.z = (unsigned)ga;                                    // global addr [31:0]
  g0.w = (unsigned)((ga >> 32) & 0x01FFFFFFu) | (2u << 30); // [56:32] | type=2
  v8i g1;
  g1[0] = (1 << 16);                                      // data_size=1 (2 B)
  g1[1] = (tensor_d0 & 0xFFFF) << 16;                     // tensor_dim0[15:0]
  g1[2] = ((tensor_d0 >> 16) & 0xFFFF) | ((tensor_d1 & 0xFFFF) << 16);
  g1[3] = ((tensor_d1 >> 16) & 0xFFFF) | ((tile_d0 & 0xFFFF) << 16);
  g1[4] = (tile_d1 & 0xFFFF);                             // tile_dim1; dim2=0
  g1[5] = (int)(stride0 & 0xFFFFFFFFll);                  // dim0_stride[31:0]
  g1[6] = (int)((stride0 >> 32) & 0xFFFFll);              // dim0_stride[47:32]
  g1[7] = 0;
  const v4i g2 = {0, 0, 0, 0};                            // 2-D: groups 2/3 unused
  const v4i g3 = {0, 0, 0, 0};
  const v8i g4 = {0, 0, 0, 0, 0, 0, 0, 0};
  __builtin_amdgcn_tensor_load_to_lds(g0, g1, g2, g3, g4, 0);
}
#endif

// ---------------------------------------------------------------------------
// Kernel 1: fused QKV projection.  One wave computes a 16x64 output tile of
// one of {Q,K,V} (blockIdx.z selects which).  Output stored bf16 in workspace
// with layout [b][h][s][d] so the attention kernel reads contiguously.
// ---------------------------------------------------------------------------
__global__ __launch_bounds__(32)
void qkv_proj_kernel(const float* __restrict__ q,
                     const float* __restrict__ k,
                     const float* __restrict__ v,
                     const float* __restrict__ Wq,
                     const float* __restrict__ Wk,
                     const float* __restrict__ Wv,
                     __bf16* __restrict__ ws) {
  const int which = blockIdx.z;
  const float* X = (which == 0) ? q  : (which == 1) ? k  : v;
  const float* W = (which == 0) ? Wq : (which == 1) ? Wk : Wv;
  __bf16* out = ws + (size_t)which * (size_t)B_ * H_ * S_ * D_;

  const int m0 = blockIdx.x * 16;     // row tile over B*S
  const int n0 = blockIdx.y * 64;     // 64-wide col group over H*D

  v8f acc[4] = {};
  for (int k0 = 0; k0 < F_; k0 += 32) {
    v16bf a = load_a_f32(X + (size_t)m0 * F_ + k0, F_);
#pragma unroll
    for (int j = 0; j < 4; ++j) {
      v16bf b = load_b_f32(W + (size_t)k0 * HD_ + n0 + 16 * j, HD_);
      acc[j] = wmma_bf16(a, b, acc[j]);
    }
  }

  const int bb = m0 / S_;
  const int ss = m0 % S_;
#pragma unroll
  for (int j = 0; j < 4; ++j) {
    const int col = n0 + 16 * j;
    const int h = col >> 6;
    const int d = col & 63;
    __bf16* ob = out + (((size_t)bb * H_ + h) * S_ + ss) * D_ + d;
    store_c_bf16(ob, D_, acc[j]);
  }
}

// ---------------------------------------------------------------------------
// Kernel 2: attention with streamed (flash-style, recompute) softmax.
// One wave per (b,h, 16-query tile).  Pass 1: online row max/sum over all
// 2048 keys (K tiles are L2-resident: 256 KB/head << 192 MB L2).  Pass 2:
// recompute scores, write normalized probs to p_attn, and accumulate P*V.
// V tiles are staged into a double-buffered LDS tile by the Tensor Data
// Mover (async, TENSORcnt), overlapping the DMA with the score WMMAs and
// turning the strided V accesses into LDS traffic.
// ---------------------------------------------------------------------------
__global__ __launch_bounds__(32)
void attention_kernel(const __bf16* __restrict__ Q,
                      const __bf16* __restrict__ K,
                      const __bf16* __restrict__ V,
                      const unsigned char* __restrict__ mask,
                      float* __restrict__ p_attn,
                      __bf16* __restrict__ AO) {
  const int q0 = blockIdx.x * 16;     // query tile
  const int bh = blockIdx.y;          // b*H + h
  const int bb = bh / H_;
  const int lane = threadIdx.x & 31;
  const int cn = lane & 15;           // C-fragment column
  const int r0 = (lane >> 4) << 3;    // C-fragment row base

  __shared__ float  stile[16][16];
  __shared__ float  rowmax[16];
  __shared__ float  rowsum[16];
  __shared__ __bf16 ptile[16][32];
  __shared__ __bf16 vtile[2][32 * D_];   // double-buffered V tiles (8 KB)

  const __bf16* Qb = Q + ((size_t)bh * S_ + q0) * D_;
  const __bf16* Kb = K + (size_t)bh * S_ * D_;
  const __bf16* Vb = V + (size_t)bh * S_ * D_;
  const unsigned char* mk = mask + (size_t)bb * S_;

  const v16bf qa0 = load_a_bf16(Qb, D_);        // d = 0..31
  const v16bf qa1 = load_a_bf16(Qb + 32, D_);   // d = 32..63
  const float scale = 0.125f;                   // 1/sqrt(64)

  if (lane < 16) { rowmax[lane] = -3.0e38f; rowsum[lane] = 0.0f; }
  __syncthreads();

  // ---- Pass 1: online row max & sum ------------------------------------
  for (int kt = 0; kt < S_; kt += 16) {
    if (kt + 16 < S_)
      __builtin_prefetch(Kb + (size_t)(kt + 16) * D_, 0, 3);
    v8f c = {};
    c = wmma_bf16(qa0, load_bT_bf16(Kb + (size_t)kt * D_, D_), c);
    c = wmma_bf16(qa1, load_bT_bf16(Kb + (size_t)kt * D_ + 32, D_), c);
#pragma unroll
    for (int r = 0; r < 8; ++r) stile[r0 + r][cn] = c[r];
    __syncthreads();
    if (lane < 16) {
      float m = rowmax[lane], l = rowsum[lane];
      for (int j = 0; j < 16; ++j) {
        const float sc = mk[kt + j] ? stile[lane][j] * scale : -1.0e9f;
        const float mn = fmaxf(m, sc);
        l = l * __expf(m - mn) + __expf(sc - mn);
        m = mn;
      }
      rowmax[lane] = m; rowsum[lane] = l;
    }
    __syncthreads();
  }

  if (lane < 16) rowsum[lane] = 1.0f / rowsum[lane];
  __syncthreads();

  // ---- Pass 2: probabilities out + P*V ---------------------------------
  v8f o[4] = {};
  float* prow = p_attn + ((size_t)bh * S_ + q0) * S_;

  // Stage a 32-key x 64-d V tile into LDS buffer `buf`.
  auto stage_v = [&](int buf, int key0) {
#if HAVE_TDM
    tdm_load_2d_bf16((unsigned)(uintptr_t)&vtile[buf][0],
                     Vb + (size_t)key0 * D_,
                     /*tile*/ D_, 32, /*tensor*/ D_, S_, /*stride0*/ (long long)D_);
#else
    const unsigned* src = (const unsigned*)(Vb + (size_t)key0 * D_);
    unsigned* dst = (unsigned*)&vtile[buf][0];
    for (int i = lane; i < (32 * D_) / 2; i += 32) dst[i] = src[i];
#endif
  };

  // Recompute scores for keys [kt, kt+32), write p_attn, fill ptile (bf16).
  auto probs_step = [&](int kt) {
#pragma unroll
    for (int half = 0; half < 2; ++half) {
      const int kk = kt + half * 16;
      v8f c = {};
      c = wmma_bf16(qa0, load_bT_bf16(Kb + (size_t)kk * D_, D_), c);
      c = wmma_bf16(qa1, load_bT_bf16(Kb + (size_t)kk * D_ + 32, D_), c);
      const bool mv = (mk[kk + cn] != 0);        // mask depends on key only
#pragma unroll
      for (int r = 0; r < 8; ++r) {
        const int row = r0 + r;
        const float sc = mv ? c[r] * scale : -1.0e9f;
        const float p  = __expf(sc - rowmax[row]) * rowsum[row];
        prow[(size_t)row * S_ + kk + cn] = p;           // p_attn (f32)
        ptile[row][half * 16 + cn] = (__bf16)p;         // stage for WMMA A
      }
    }
  };

  // Consume ptile and LDS-resident V tile with 4 WMMAs.
  auto pv_step = [&](int buf) {
    __syncthreads();
    const v16bf pa = load_a_bf16(&ptile[0][0], 32);     // 16x32 prob tile
#pragma unroll
    for (int j = 0; j < 4; ++j) {
      v16bf vb = load_b_bf16(&vtile[buf][16 * j], D_);
      o[j] = wmma_bf16(pa, vb, o[j]);
    }
    __syncthreads();
  };

  stage_v(0, 0);                                        // prologue DMA
  for (int kt = 0; kt + 32 < S_; kt += 32) {            // steady state
    const int buf = (kt >> 5) & 1;
    stage_v(buf ^ 1, kt + 32);                          // DMA next tile
    probs_step(kt);                                     // overlaps the DMA
#if HAVE_TDM
    __builtin_amdgcn_s_wait_tensorcnt((short)1);        // current tile ready
#endif
    pv_step(buf);
  }
  {                                                     // peeled last tile
    const int kt = S_ - 32;
    probs_step(kt);
#if HAVE_TDM
    __builtin_amdgcn_s_wait_tensorcnt((short)0);
#endif
    pv_step((kt >> 5) & 1);
  }

  __bf16* AOb = AO + ((size_t)bh * S_ + q0) * D_;
#pragma unroll
  for (int j = 0; j < 4; ++j) store_c_bf16(AOb + 16 * j, D_, o[j]);
}

// ---------------------------------------------------------------------------
// Kernel 3: output projection.  AO is [b][h][s][d]; each 32-wide K chunk of
// the logical [b][s][h*d] matrix lies inside a single head (64 % 32 == 0),
// so A-fragments load contiguously.
// ---------------------------------------------------------------------------
__global__ __launch_bounds__(32)
void out_proj_kernel(const __bf16* __restrict__ AO,
                     const float* __restrict__ Wo,
                     float* __restrict__ out) {
  const int m0 = blockIdx.x * 16;     // row tile over B*S
  const int n0 = blockIdx.y * 64;     // 64-wide col group over F
  const int bb = m0 / S_;
  const int ss = m0 % S_;

  v8f acc[4] = {};
  for (int k0 = 0; k0 < HD_; k0 += 32) {
    const int h = k0 >> 6;
    const int d = k0 & 63;
    v16bf a = load_a_bf16(AO + (((size_t)bb * H_ + h) * S_ + ss) * D_ + d, D_);
#pragma unroll
    for (int j = 0; j < 4; ++j) {
      v16bf w = load_b_f32(Wo + (size_t)k0 * F_ + n0 + 16 * j, F_);
      acc[j] = wmma_bf16(a, w, acc[j]);
    }
  }
  float* ob = out + ((size_t)bb * S_ + ss) * F_ + n0;
#pragma unroll
  for (int j = 0; j < 4; ++j) store_c_f32(ob + 16 * j, F_, acc[j]);
}

// ---------------------------------------------------------------------------
extern "C" void kernel_launch(void* const* d_in, const int* in_sizes, int n_in,
                              void* d_out, int out_size, void* d_ws, size_t ws_size,
                              hipStream_t stream) {
  (void)in_sizes; (void)n_in; (void)out_size; (void)ws_size;

  const float* query  = (const float*)d_in[0];
  const float* keys   = (const float*)d_in[1];
  const float* values = (const float*)d_in[2];
  const unsigned char* mask = (const unsigned char*)d_in[3];
  const float* Wq = (const float*)d_in[4];
  const float* Wk = (const float*)d_in[5];
  const float* Wv = (const float*)d_in[6];
  const float* Wo = (const float*)d_in[7];

  float* out    = (float*)d_out;                       // (B,S,F)
  float* p_attn = out + (size_t)B_ * S_ * F_;          // (B,H,S,S)

  // Workspace (bf16): Q | K | V | AO, each B*H*S*D elems -> 64 MiB total.
  __bf16* ws = (__bf16*)d_ws;
  const size_t qkv = (size_t)B_ * H_ * S_ * D_;
  __bf16* Qw = ws;
  __bf16* Kw = ws + qkv;
  __bf16* Vw = ws + 2 * qkv;
  __bf16* AO = ws + 3 * qkv;

  const dim3 blk(32, 1, 1);

  qkv_proj_kernel<<<dim3((B_ * S_) / 16, HD_ / 64, 3), blk, 0, stream>>>(
      query, keys, values, Wq, Wk, Wv, ws);

  attention_kernel<<<dim3(S_ / 16, B_ * H_, 1), blk, 0, stream>>>(
      Qw, Kw, Vw, mask, p_attn, AO);

  out_proj_kernel<<<dim3((B_ * S_) / 16, F_ / 64, 1), blk, 0, stream>>>(
      AO, Wo, out);
}
